// IntentNERModel_41085657153956
// MI455X (gfx1250) — matrix operand order repaired and checked by reading, not compiled, not run on hardware
//
#include <hip/hip_runtime.h>
#include <hip/hip_bf16.h>

// Problem constants (from reference)
#define BB 4096
#define TT 20
#define VV 100000
#define DD 300
#define HH 128
#define NI 15
#define NE 8
#define DK 320          // K padded to multiple of 32
#define NX 768          // combined N: [fw_gate 256 | fw_cand 128 | bw_gate 256 | bw_cand 128]

typedef __attribute__((ext_vector_type(16))) _Float16 v16h;
typedef __attribute__((ext_vector_type(8)))  float    v8f;
typedef __attribute__((ext_vector_type(4)))  float    v4f;
typedef __attribute__((ext_vector_type(4)))  _Float16 v4h;

// ---- WMMA fragment loaders (wave32, 16x16x32 f16) ----
// A (16x32 tile from row-major f16, stride ld): lane&15 = row M, lane>>4 -> K offset 0/16 (16 contiguous K).
__device__ inline v16h frag_a(const _Float16* base, int ld) {
    int lane = threadIdx.x & 31;
    int row  = lane & 15;
    int kb   = (lane >> 4) << 4;            // 0 or 16
    const _Float16* p = base + row * ld + kb;
    v16h f;
#pragma unroll
    for (int v = 0; v < 4; ++v) {
        f[2*v]     = p[2*v];
        f[2*v + 1] = p[2*v + 1];
        f[8 + 2*v]     = p[8 + 2*v];
        f[8 + 2*v + 1] = p[8 + 2*v + 1];
    }
    return f;
}
// B fragment from PRE-PACKED storage: 32 lanes x 16 halfs, lane-contiguous (32B/lane).
// Packed element e of lane L corresponds to K = (L>>4)*16 + e, N = L&15.
__device__ inline v16h frag_b_packed(const _Float16* frag_base) {
    int lane = threadIdx.x & 31;
    return *(const v16h*)(frag_base + (size_t)lane * 16);
}

// ---------------- seqlen ----------------
__global__ void seqlen_kernel(const int* __restrict__ x, int* __restrict__ seqlen) {
    int b = blockIdx.x * blockDim.x + threadIdx.x;
    if (b >= BB) return;
    int c = 0;
#pragma unroll
    for (int t = 0; t < TT; ++t) c += (x[b * TT + t] != 0);
    seqlen[b] = c;
}

// ---------------- zero fill ----------------
__global__ void zero_kernel(float* __restrict__ p, int n) {
    int i = blockIdx.x * blockDim.x + threadIdx.x;
    if (i < n) p[i] = 0.0f;
}

// ---------------- weight prep: f32 -> f16 WMMA-B-fragment packs ----------------
// wx : 10 ktiles x 48 ntiles fragments (zero pad k>=300); whg: [2][4][16]; whc: [2][4][8].
// Fragment = 512 halfs: lane L x element e; K = ktile*32 + (L>>4)*16 + e, N = ntile*16 + (L&15).
__global__ void prep_weights_kernel(const float* __restrict__ fwg, const float* __restrict__ fwc,
                                    const float* __restrict__ bwg, const float* __restrict__ bwc,
                                    _Float16* __restrict__ wx, _Float16* __restrict__ whg,
                                    _Float16* __restrict__ whc) {
    int idx = blockIdx.x * blockDim.x + threadIdx.x;
    const int NWX = 10 * 48 * 512;           // 245760
    if (idx < NWX) {
        int frag = idx >> 9, within = idx & 511;
        int lane = within >> 4, e = within & 15;
        int ktile = frag / 48, ntile = frag % 48;
        int k = ktile * 32 + ((lane >> 4) << 4) + e;
        int n = ntile * 16 + (lane & 15);
        float v = 0.0f;
        if (k < DD) {
            if      (n < 256) v = fwg[k * 256 + n];
            else if (n < 384) v = fwc[k * 128 + (n - 256)];
            else if (n < 640) v = bwg[k * 256 + (n - 384)];
            else              v = bwc[k * 128 + (n - 640)];
        }
        wx[idx] = (_Float16)v;
        return;
    }
    int i2 = idx - NWX;                      // whg: 2 * 4*16*512 = 65536
    if (i2 < 2 * 4 * 16 * 512) {
        int dir = i2 >> 15, rem = i2 & 32767;
        int frag = rem >> 9, within = rem & 511;
        int lane = within >> 4, e = within & 15;
        int ktile = frag >> 4, ntile = frag & 15;
        int k = ktile * 32 + ((lane >> 4) << 4) + e;
        int n = ntile * 16 + (lane & 15);
        const float* src = dir ? bwg : fwg;
        whg[i2] = (_Float16)src[(DD + k) * 256 + n];
        return;
    }
    int i3 = i2 - 2 * 4 * 16 * 512;          // whc: 2 * 4*8*512 = 32768
    if (i3 < 2 * 4 * 8 * 512) {
        int dir = i3 >> 14, rem = i3 & 16383;
        int frag = rem >> 9, within = rem & 511;
        int lane = within >> 4, e = within & 15;
        int ktile = frag >> 3, ntile = frag & 7;
        int k = ktile * 32 + ((lane >> 4) << 4) + e;
        int n = ntile * 16 + (lane & 15);
        const float* src = dir ? bwc : fwc;
        whc[i3] = (_Float16)src[(DD + k) * 128 + n];
    }
}

// ---------------- fused embedding-gather + x-projection GEMM ----------------
// (B*T x 300) @ (300 x 768) -> xp (B*T x 768) f32, bias folded in.
// Block: 256 threads (8 waves), M = 32 rows (2 A tiles) x full 768 N.
// Each B fragment feeds 2 WMMAs (M reuse) -> ~1.2 VMEM b128 per WMMA.
__global__ __launch_bounds__(256) void xproj_kernel(
        const int* __restrict__ x, const float* __restrict__ emb,
        const _Float16* __restrict__ wx,
        const float* __restrict__ fw_gb, const float* __restrict__ fw_cb,
        const float* __restrict__ bw_gb, const float* __restrict__ bw_cb,
        float* __restrict__ xp) {
    __shared__ _Float16 lds_a[32 * DK];      // 20 KB
    int m0  = blockIdx.x * 32;
    int tid = threadIdx.x;
    // cooperative gather of 32 embedding rows (float4 vectorized: 300 = 75*4),
    // convert to f16, pad K to 320 (chunks 75..79 zero)
    for (int idx = tid; idx < 32 * (DK / 4); idx += 256) {
        int r = idx / (DK / 4), c4 = idx - r * (DK / 4);
        int id = x[m0 + r];
        v4f v = {};
        if (c4 < DD / 4) v = *(const v4f*)(emb + (size_t)id * DD + c4 * 4);
        v4h h;
#pragma unroll
        for (int e = 0; e < 4; ++e) h[e] = (_Float16)v[e];
        *(v4h*)(lds_a + r * DK + c4 * 4) = h;
    }
    __syncthreads();

    int wave = tid >> 5;
    v8f acc0[6] = {};
    v8f acc1[6] = {};
    for (int kt = 0; kt < DK / 32; ++kt) {
        v16h a0 = frag_a(lds_a + kt * 32, DK);
        v16h a1 = frag_a(lds_a + 16 * DK + kt * 32, DK);
#pragma unroll
        for (int j = 0; j < 6; ++j) {
            v16h b = frag_b_packed(wx + (size_t)(kt * 48 + wave * 6 + j) * 512);
            acc0[j] = __builtin_amdgcn_wmma_f32_16x16x32_f16(
                false, a0, false, b, (short)0, acc0[j], false, false);
            acc1[j] = __builtin_amdgcn_wmma_f32_16x16x32_f16(
                false, a1, false, b, (short)0, acc1[j], false, false);
        }
    }
    // epilogue: + bias, store f32
    int lane  = tid & 31;
    int colN  = lane & 15;
    int rbase = (lane < 16) ? 0 : 8;
#pragma unroll
    for (int j = 0; j < 6; ++j) {
        int n = (wave * 6 + j) * 16 + colN;
        float bias;
        if      (n < 256) bias = fw_gb[n];
        else if (n < 384) bias = fw_cb[n - 256];
        else if (n < 640) bias = bw_gb[n - 384];
        else              bias = bw_cb[n - 640];
#pragma unroll
        for (int i = 0; i < 8; ++i) {
            int row = m0 + rbase + i;
            xp[(size_t)row * NX + n]        = acc0[j][i] + bias;
            xp[(size_t)(row + 16) * NX + n] = acc1[j][i] + bias;
        }
    }
}

// ---------------- one GRU timestep (both directions via blockIdx.y) ----------------
// Per block: 16 batch rows. gates = sigmoid(h@Whg + xp_gate); c = tanh((r*h)@Whc + xp_cand);
// h' = u*h + (1-u)*c (masked). bw writes out_bw at reversed position p = seqlen-1-t.
__global__ __launch_bounds__(256) void gru_step_kernel(
        int t, const int* __restrict__ seqlen, const float* __restrict__ xp,
        const _Float16* __restrict__ whg, const _Float16* __restrict__ whc,
        float* __restrict__ hbuf, float* __restrict__ out_bw) {
    __shared__ float    lds_h [16 * HH];         // 8 KB
    __shared__ _Float16 lds_hf[16 * HH];         // 4 KB
    __shared__ float    lds_g [16 * 2 * HH];     // 16 KB
    __shared__ _Float16 lds_rh[16 * HH];         // 4 KB

    int dir = blockIdx.y;
    int b0  = blockIdx.x * 16;
    int tid = threadIdx.x;
    float* hdir = hbuf + (size_t)dir * BB * HH;

    for (int idx = tid; idx < 16 * HH / 4; idx += 256) {
        int r = idx >> 5, c4 = idx & 31;
        v4f hv = *(const v4f*)(hdir + (size_t)(b0 + r) * HH + c4 * 4);
        *(v4f*)(lds_h + r * HH + c4 * 4) = hv;
        v4h hh;
#pragma unroll
        for (int e = 0; e < 4; ++e) hh[e] = (_Float16)hv[e];
        *(v4h*)(lds_hf + r * HH + c4 * 4) = hh;
    }
    __syncthreads();

    int wave = tid >> 5, lane = tid & 31;
    int colN = lane & 15, rbase = (lane < 16) ? 0 : 8;
    int xgate = dir ? 384 : 0;
    int xcand = dir ? 640 : 256;

    // ---- gate matmul: (16x128)@(128x256); wave owns 2 N-tiles ----
    {
        v8f acc[2] = {};
        const _Float16* wg = whg + (size_t)dir * (4 * 16 * 512);
        for (int kt = 0; kt < 4; ++kt) {
            v16h a = frag_a(lds_hf + kt * 32, HH);
#pragma unroll
            for (int j = 0; j < 2; ++j) {
                v16h b = frag_b_packed(wg + (size_t)(kt * 16 + wave * 2 + j) * 512);
                acc[j] = __builtin_amdgcn_wmma_f32_16x16x32_f16(
                    false, a, false, b, (short)0, acc[j], false, false);
            }
        }
#pragma unroll
        for (int j = 0; j < 2; ++j) {
            int n = (wave * 2 + j) * 16 + colN;
#pragma unroll
            for (int i = 0; i < 8; ++i) {
                int r  = rbase + i;
                int b  = b0 + r;
                int sl = seqlen[b];
                int p  = (t < sl) ? (dir ? (sl - 1 - t) : t) : t;
                float g = acc[j][i] + xp[((size_t)b * TT + p) * NX + xgate + n];
                lds_g[r * 2 * HH + n] = 1.0f / (1.0f + __expf(-g));
            }
        }
    }
    __syncthreads();
    // rh = r * h (f16), vectorized
    for (int idx = tid; idx < 16 * HH / 4; idx += 256) {
        int r = idx >> 5, c4 = idx & 31;
        v4f rv = *(const v4f*)(lds_g + r * 2 * HH + c4 * 4);
        v4f hv = *(const v4f*)(lds_h + r * HH + c4 * 4);
        v4h o;
#pragma unroll
        for (int e = 0; e < 4; ++e) o[e] = (_Float16)(rv[e] * hv[e]);
        *(v4h*)(lds_rh + r * HH + c4 * 4) = o;
    }
    __syncthreads();
    // ---- cand matmul: (16x128)@(128x128); wave owns 1 N-tile ----
    {
        v8f acc = {};
        const _Float16* wc = whc + (size_t)dir * (4 * 8 * 512);
        for (int kt = 0; kt < 4; ++kt) {
            v16h a = frag_a(lds_rh + kt * 32, HH);
            v16h b = frag_b_packed(wc + (size_t)(kt * 8 + wave) * 512);
            acc = __builtin_amdgcn_wmma_f32_16x16x32_f16(
                false, a, false, b, (short)0, acc, false, false);
        }
#pragma unroll
        for (int i = 0; i < 8; ++i) {
            int r  = rbase + i;
            int b  = b0 + r;
            int c  = wave * 16 + colN;
            int sl = seqlen[b];
            bool valid = (t < sl);
            int p = valid ? (dir ? (sl - 1 - t) : t) : t;
            float cv   = tanhf(acc[i] + xp[((size_t)b * TT + p) * NX + xcand + c]);
            float u    = lds_g[r * 2 * HH + HH + c];
            float hold = lds_h[r * HH + c];
            float hn   = u * hold + (1.0f - u) * cv;
            hdir[(size_t)b * HH + c] = valid ? hn : hold;
            if (dir == 1 && valid) out_bw[((size_t)b * TT + p) * HH + c] = hn;
        }
    }
}

// ---------------- heads ----------------
__global__ void intent_kernel(const float* __restrict__ hfw, const float* __restrict__ w_int,
                              const float* __restrict__ b_int, float* __restrict__ out) {
    int idx = blockIdx.x * blockDim.x + threadIdx.x;
    if (idx >= BB * NI) return;
    int b = idx / NI, i = idx % NI;
    float s = b_int[i];
#pragma unroll 8
    for (int k = 0; k < HH; ++k) s += hfw[(size_t)b * HH + k] * w_int[k * NI + i];
    out[idx] = s;
}

__global__ void ner_kernel(const float* __restrict__ out_bw, const float* __restrict__ intent,
                           const float* __restrict__ w_ent, const float* __restrict__ b_ent,
                           float* __restrict__ out) {
    int idx = blockIdx.x * blockDim.x + threadIdx.x;
    if (idx >= BB * TT * NE) return;
    int e = idx % NE, bt = idx / NE, b = bt / TT;
    float s = b_ent[e];
#pragma unroll 8
    for (int k = 0; k < HH; ++k) s += out_bw[(size_t)bt * HH + k] * w_ent[k * NE + e];
#pragma unroll
    for (int i = 0; i < NI; ++i) s += intent[b * NI + i] * w_ent[(HH + i) * NE + e];
    out[idx] = s;
}

// ---------------- launcher ----------------
extern "C" void kernel_launch(void* const* d_in, const int* in_sizes, int n_in,
                              void* d_out, int out_size, void* d_ws, size_t ws_size,
                              hipStream_t stream) {
    const int*   x      = (const int*)  d_in[0];
    const float* emb    = (const float*)d_in[1];
    const float* fwg    = (const float*)d_in[2];
    const float* fwgb   = (const float*)d_in[3];
    const float* fwc    = (const float*)d_in[4];
    const float* fwcb   = (const float*)d_in[5];
    const float* bwg    = (const float*)d_in[6];
    const float* bwgb   = (const float*)d_in[7];
    const float* bwc    = (const float*)d_in[8];
    const float* bwcb   = (const float*)d_in[9];
    const float* w_int  = (const float*)d_in[10];
    const float* b_int  = (const float*)d_in[11];
    const float* w_ent  = (const float*)d_in[12];
    const float* b_ent  = (const float*)d_in[13];
    float* out = (float*)d_out;

    // workspace layout
    char* ws = (char*)d_ws;
    size_t off = 0;
    auto take = [&](size_t bytes) { char* p = ws + off; off = (off + bytes + 255) & ~(size_t)255; return p; };
    int*      seqlen = (int*)      take((size_t)BB * 4);
    float*    xp     = (float*)    take((size_t)BB * TT * NX * 4);
    _Float16* wx     = (_Float16*) take((size_t)10 * 48 * 512 * 2);
    _Float16* whg    = (_Float16*) take((size_t)2 * 4 * 16 * 512 * 2);
    _Float16* whc    = (_Float16*) take((size_t)2 * 4 * 8 * 512 * 2);
    float*    hbuf   = (float*)    take((size_t)2 * BB * HH * 4);
    float*    obw    = (float*)    take((size_t)BB * TT * HH * 4);

    seqlen_kernel<<<(BB + 255) / 256, 256, 0, stream>>>(x, seqlen);

    int prep_n = 10 * 48 * 512 + 2 * 4 * 16 * 512 + 2 * 4 * 8 * 512;
    prep_weights_kernel<<<(prep_n + 255) / 256, 256, 0, stream>>>(fwg, fwc, bwg, bwc, wx, whg, whc);

    zero_kernel<<<(2 * BB * HH + 255) / 256, 256, 0, stream>>>(hbuf, 2 * BB * HH);
    zero_kernel<<<(BB * TT * HH + 255) / 256, 256, 0, stream>>>(obw, BB * TT * HH);

    xproj_kernel<<<(BB * TT) / 32, 256, 0, stream>>>(x, emb, wx, fwgb, fwcb, bwgb, bwcb, xp);

    for (int t = 0; t < TT; ++t) {
        gru_step_kernel<<<dim3(BB / 16, 2), 256, 0, stream>>>(t, seqlen, xp, whg, whc, hbuf, obw);
    }

    // forward final h == last_fw (hbuf dir 0)
    intent_kernel<<<(BB * NI + 255) / 256, 256, 0, stream>>>(hbuf, w_int, b_int, out);
    ner_kernel<<<(BB * TT * NE + 255) / 256, 256, 0, stream>>>(obw, out, w_ent, b_ent, out + BB * NI);
}
